// Pointnet2SSG_seg_feature_26164940767538
// MI455X (gfx1250) — matrix operand order repaired and checked by reading, not compiled, not run on hardware
//
#include <hip/hip_runtime.h>

typedef __bf16 bf16;
typedef __attribute__((ext_vector_type(16))) __bf16 v16bf;
typedef __attribute__((ext_vector_type(8)))  __bf16 v8bf;
typedef __attribute__((ext_vector_type(8)))  float  v8f;

#define DEV __device__ __forceinline__

// ---------------- WMMA fragment helpers (CDNA5 16x16x32 bf16) ----------------
// A-matrix 16x32 (MxK): lane half h holds K = h*8..h*8+7 (elems 0-7) and
// K = 16+h*8 .. 16+h*8+7 (elems 8-15), row M = lane&15.
DEV v16bf load_a_frag(const bf16* base, int ld, int lane) {
  int r = lane & 15, h = lane >> 4;
  const bf16* p = base + (size_t)r * ld + h * 8;
  v8bf lo = *(const v8bf*)(p);
  v8bf hi = *(const v8bf*)(p + 16);
  v16bf f;
#pragma unroll
  for (int i = 0; i < 8; ++i) { f[i] = lo[i]; f[i + 8] = hi[i]; }
  return f;
}
// B-matrix 32x16 (KxN), column-major per lane: lane = col n, half h holds
// K = h*8..h*8+7 (elems 0-7) and K = 16+h*8 .. 16+h*8+7 (elems 8-15).
DEV v16bf load_b_frag(const bf16* base, int ld, int lane) {
  int n = lane & 15, h = lane >> 4;
  const bf16* p = base + (size_t)n * ld + h * 8;
  v8bf lo = *(const v8bf*)(p);
  v8bf hi = *(const v8bf*)(p + 16);
  v16bf f;
#pragma unroll
  for (int i = 0; i < 8; ++i) { f[i] = lo[i]; f[i + 8] = hi[i]; }
  return f;
}
DEV v8f wmma_bf16(v16bf a, v16bf b, v8f c) {
  return __builtin_amdgcn_wmma_f32_16x16x32_bf16(false, a, false, b, (short)0, c,
                                                 false, false);
}

// ---------------- weight packing: fp32 [cout][cin] -> bf16 [cout][kpad] -----
__global__ void pack_weight(const float* __restrict__ W, bf16* __restrict__ D,
                            int cout, int cin, int kp) {
  int t = blockIdx.x * blockDim.x + threadIdx.x;
  if (t >= cout * kp) return;
  int o = t / kp, k = t % kp;
  D[t] = (k < cin) ? (bf16)W[(size_t)o * cin + k] : (bf16)0.f;
}

// ---------------- farthest point sampling (one block per batch) -------------
// Point cloud staged in LDS planes: the 512 dependent iterations then run
// entirely out of LDS (ds_load latency instead of VMEM round trips).
__global__ void __launch_bounds__(1024) fps_kernel(
    const float* __restrict__ pts, int pstride, int n, int npoint,
    int* __restrict__ oidx, float* __restrict__ oxyz) {
  extern __shared__ float sm[];
  float* sx = sm;
  float* sy = sm + n;
  float* sz = sm + 2 * n;
  __shared__ float rv[32];
  __shared__ int   ri[32];
  __shared__ int   sfar;
  int b = blockIdx.x;
  const float* P = pts + (size_t)b * n * pstride;
  const int T = blockDim.x;
  int tid = threadIdx.x;
  for (int i = tid; i < n; i += T) {
    sx[i] = P[(size_t)i * pstride + 0];
    sy[i] = P[(size_t)i * pstride + 1];
    sz[i] = P[(size_t)i * pstride + 2];
  }
  float d[8];
  int cnt = (n + T - 1) / T;
#pragma unroll
  for (int i = 0; i < 8; ++i) d[i] = 1e10f;
  if (tid == 0) sfar = 0;
  __syncthreads();
  for (int it = 0; it < npoint; ++it) {
    int far = sfar;
    float cx = sx[far], cy = sy[far], cz = sz[far];
    if (tid == 0) {
      oidx[(size_t)b * npoint + it] = far;
      oxyz[((size_t)b * npoint + it) * 3 + 0] = cx;
      oxyz[((size_t)b * npoint + it) * 3 + 1] = cy;
      oxyz[((size_t)b * npoint + it) * 3 + 2] = cz;
    }
    float bv = -1.f; int bi = 0x7fffffff;
    for (int i = 0; i < cnt; ++i) {
      int p = tid + i * T;
      if (p < n) {
        float dx = sx[p] - cx, dy = sy[p] - cy, dz = sz[p] - cz;
        float dd = dx * dx + dy * dy + dz * dz;
        float nd = fminf(d[i], dd);
        d[i] = nd;
        if (nd > bv || (nd == bv && p < bi)) { bv = nd; bi = p; }
      }
    }
    for (int off = 16; off; off >>= 1) {
      float ov = __shfl_xor(bv, off, 32);
      int   oi = __shfl_xor(bi, off, 32);
      if (ov > bv || (ov == bv && oi < bi)) { bv = ov; bi = oi; }
    }
    int w = tid >> 5;
    if ((tid & 31) == 0) { rv[w] = bv; ri[w] = bi; }
    __syncthreads();
    if (w == 0) {
      int nw = T >> 5;
      bv = (tid < nw) ? rv[tid] : -1.f;
      bi = (tid < nw) ? ri[tid] : 0x7fffffff;
      for (int off = 16; off; off >>= 1) {
        float ov = __shfl_xor(bv, off, 32);
        int   oi = __shfl_xor(bi, off, 32);
        if (ov > bv || (ov == bv && oi < bi)) { bv = ov; bi = oi; }
      }
      if (tid == 0) sfar = bi;
    }
    __syncthreads();
  }
}

// ---------------- ball query: one wave32 per centroid -----------------------
// Batch point cloud staged once per block in LDS (8 waves share it).
__global__ void __launch_bounds__(256) ball_query(
    const float* __restrict__ pts, int pstride,
    const float* __restrict__ centers, int Bn, int n, int nc, float r2,
    int* __restrict__ out) {
  extern __shared__ float sm[];
  float* sx = sm;
  float* sy = sm + n;
  float* sz = sm + 2 * n;
  int gw0 = blockIdx.x * (blockDim.x >> 5);
  int b = gw0 / nc;  // uniform per block: nc is a multiple of waves/block
  const float* P = pts + (size_t)b * n * pstride;
  for (int i = threadIdx.x; i < n; i += blockDim.x) {
    sx[i] = P[(size_t)i * pstride + 0];
    sy[i] = P[(size_t)i * pstride + 1];
    sz[i] = P[(size_t)i * pstride + 2];
  }
  __syncthreads();
  int lane = threadIdx.x & 31;
  int gw = gw0 + (threadIdx.x >> 5);
  if (gw >= Bn * nc) return;
  float cx = centers[(size_t)gw * 3 + 0];
  float cy = centers[(size_t)gw * 3 + 1];
  float cz = centers[(size_t)gw * 3 + 2];
  int* o = out + ((size_t)gw << 6);
  int count = 0, first = n - 1;
  unsigned lmask = (1u << lane) - 1u;
  for (int base = 0; base < n && count < 64; base += 32) {
    int p = base + lane;
    bool in = false;
    if (p < n) {
      float dx = sx[p] - cx, dy = sy[p] - cy, dz = sz[p] - cz;
      in = (dx * dx + dy * dy + dz * dz) <= r2;
    }
    unsigned m = (unsigned)__ballot(in);
    if (count == 0 && m) first = base + __ffs((int)m) - 1;
    if (in) {
      int pos = count + __popc(m & lmask);
      if (pos < 64) o[pos] = p;
    }
    count += __popc(m);
  }
  if (count > 64) count = 64;
  for (int k = count + lane; k < 64; k += 32) o[k] = first;
}

// ---------------- LDS-resident MLP layer (M rows x N cols, WMMA) ------------
DEV void mlp_layer_lds(const bf16* in, int ldin, int Mrows, const bf16* Wg,
                       int Kp, int Nn, const float* sc, const float* bi,
                       bf16* outp, int ldout, int lane, int wave) {
  int mTiles = Mrows >> 4, nTiles = Nn >> 4;
  int tiles = mTiles * nTiles;
  for (int tt = wave; tt < tiles; tt += 8) {
    int mt = tt % mTiles, nt = tt / mTiles;
    v8f acc = {};
    for (int k = 0; k < Kp; k += 32) {
      v16bf a = load_a_frag(in + (size_t)(mt << 4) * ldin + k, ldin, lane);
      v16bf b = load_b_frag(Wg + (size_t)(nt << 4) * Kp + k, Kp, lane);
      acc = wmma_bf16(a, b, acc);
    }
    int n = (nt << 4) + (lane & 15);
    float ss = sc[n], bb = bi[n];
    int h = lane >> 4;
#pragma unroll
    for (int v = 0; v < 8; ++v) {
      float x = acc[v] * ss + bb;
      x = x > 0.f ? x : 0.f;
      outp[(size_t)((mt << 4) + (h << 3) + v) * ldout + n] = (bf16)x;
    }
  }
}

// ---------------- fused SA1: gather + MLP[32->64->64->128] + maxpool --------
__global__ void __launch_bounds__(256) sa1_fused(
    const float* __restrict__ pc, const int* __restrict__ gidx,
    const float* __restrict__ cxyz, const bf16* __restrict__ W1,
    const float* __restrict__ s1, const float* __restrict__ b1,
    const bf16* __restrict__ W2, const float* __restrict__ s2,
    const float* __restrict__ b2, const bf16* __restrict__ W3,
    const float* __restrict__ s3, const float* __restrict__ b3,
    bf16* __restrict__ l1f) {
  __shared__ __align__(16) bf16 sIn[64 * 32];
  __shared__ __align__(16) bf16 sA[64 * 64];
  __shared__ __align__(16) bf16 sB[64 * 64];
  __shared__ __align__(16) bf16 sC[64 * 128];
  int c = blockIdx.x, b = blockIdx.y;
  int t = threadIdx.x, lane = t & 31, wave = t >> 5;
  size_t cix = (size_t)b * gridDim.x + c;
  for (int i = t; i < 64 * 32; i += 256) sIn[i] = (bf16)0.f;
  __syncthreads();
  float cx = cxyz[cix * 3 + 0], cy = cxyz[cix * 3 + 1], cz = cxyz[cix * 3 + 2];
  if (t < 64) {
    int idx = gidx[(cix << 6) + t];
    const float* pr = pc + ((size_t)b * 8192 + idx) * 6;
    bf16* row = sIn + t * 32;
    row[0] = (bf16)(pr[0] - cx); row[1] = (bf16)(pr[1] - cy);
    row[2] = (bf16)(pr[2] - cz);
    row[3] = (bf16)pr[3]; row[4] = (bf16)pr[4]; row[5] = (bf16)pr[5];
  }
  __syncthreads();
  mlp_layer_lds(sIn, 32, 64, W1, 32, 64, s1, b1, sA, 64, lane, wave);
  __syncthreads();
  mlp_layer_lds(sA, 64, 64, W2, 64, 64, s2, b2, sB, 64, lane, wave);
  __syncthreads();
  mlp_layer_lds(sB, 64, 64, W3, 64, 128, s3, b3, sC, 128, lane, wave);
  __syncthreads();
  if (t < 128) {
    float m = -1e30f;
    for (int r = 0; r < 64; ++r) {
      float v = (float)sC[r * 128 + t];
      m = v > m ? v : m;
    }
    l1f[(cix << 7) + t] = (bf16)m;
  }
}

// ---------------- fused SA2: gather + MLP[160->128->128->256] + maxpool -----
__global__ void __launch_bounds__(256) sa2_fused(
    const float* __restrict__ l1xyz, const bf16* __restrict__ l1f,
    const int* __restrict__ gidx, const float* __restrict__ cxyz,
    const bf16* __restrict__ W1, const float* __restrict__ s1,
    const float* __restrict__ b1, const bf16* __restrict__ W2,
    const float* __restrict__ s2, const float* __restrict__ b2,
    const bf16* __restrict__ W3, const float* __restrict__ s3,
    const float* __restrict__ b3, bf16* __restrict__ l2f) {
  __shared__ __align__(16) bf16 sIn[64 * 160];
  __shared__ __align__(16) bf16 sA[64 * 128];
  __shared__ __align__(16) bf16 sB[64 * 128];
  __shared__ __align__(16) bf16 sC[64 * 256];
  int c = blockIdx.x, b = blockIdx.y;
  int t = threadIdx.x, lane = t & 31, wave = t >> 5;
  size_t cix = (size_t)b * gridDim.x + c;
  for (int i = t; i < 64 * 160; i += 256) sIn[i] = (bf16)0.f;
  __syncthreads();
  float cx = cxyz[cix * 3 + 0], cy = cxyz[cix * 3 + 1], cz = cxyz[cix * 3 + 2];
  int j = t >> 2, q = t & 3;
  int idx = gidx[(cix << 6) + j];
  const bf16* fr = l1f + (((size_t)b * 512 + idx) << 7);
  bf16* row = sIn + j * 160;
  if (q == 0) {
    const float* pr = l1xyz + ((size_t)b * 512 + idx) * 3;
    row[0] = (bf16)(pr[0] - cx); row[1] = (bf16)(pr[1] - cy);
    row[2] = (bf16)(pr[2] - cz);
  }
  for (int k = q * 32; k < q * 32 + 32; ++k) row[3 + k] = fr[k];
  __syncthreads();
  mlp_layer_lds(sIn, 160, 64, W1, 160, 128, s1, b1, sA, 128, lane, wave);
  __syncthreads();
  mlp_layer_lds(sA, 128, 64, W2, 128, 128, s2, b2, sB, 128, lane, wave);
  __syncthreads();
  mlp_layer_lds(sB, 128, 64, W3, 128, 256, s3, b3, sC, 256, lane, wave);
  __syncthreads();
  {
    float m = -1e30f;
    for (int r = 0; r < 64; ++r) {
      float v = (float)sC[r * 256 + t];
      m = v > m ? v : m;
    }
    l2f[(cix << 8) + t] = (bf16)m;
  }
}

// ---------------- generic GEMM: out = relu(A @ W^T * s + b) -----------------
// A:[M,lda] bf16, W:[N,ldw] bf16, each wave owns a 16x64 strip of C.
// store_mode 0: bf16 row-major [M,ldo]; 1: f32 transposed [B][N][rpb], m=b*rpb+r
__global__ void __launch_bounds__(256) gemm_bf16_wmma(
    const bf16* __restrict__ A, int lda, const bf16* __restrict__ W, int ldw,
    const float* __restrict__ sc, const float* __restrict__ bi,
    void* __restrict__ outp, int ldo, int M, int Nn, int K, int store_mode,
    int rpb) {
  int lane = threadIdx.x & 31, wave = threadIdx.x >> 5;
  int mt = blockIdx.x * 8 + wave;
  if (mt * 16 >= M) return;
  int n0 = blockIdx.y * 64;
  v8f a0 = {}, a1 = {}, a2 = {}, a3 = {};
  const bf16* Arow = A + (size_t)mt * 16 * lda;
  for (int k = 0; k < K; k += 32) {
    if (k + 32 < K)  // hide next A slab's latency (global_prefetch_b8)
      __builtin_prefetch(Arow + (size_t)(lane & 15) * lda + k + 32, 0, 1);
    v16bf af = load_a_frag(Arow + k, lda, lane);
    v16bf b0 = load_b_frag(W + (size_t)(n0 + 0) * ldw + k, ldw, lane);
    v16bf b1 = load_b_frag(W + (size_t)(n0 + 16) * ldw + k, ldw, lane);
    v16bf b2 = load_b_frag(W + (size_t)(n0 + 32) * ldw + k, ldw, lane);
    v16bf b3 = load_b_frag(W + (size_t)(n0 + 48) * ldw + k, ldw, lane);
    a0 = wmma_bf16(af, b0, a0);
    a1 = wmma_bf16(af, b1, a1);
    a2 = wmma_bf16(af, b2, a2);
    a3 = wmma_bf16(af, b3, a3);
  }
  int h = lane >> 4, nl = lane & 15;
#pragma unroll
  for (int tq = 0; tq < 4; ++tq) {
    v8f acc = tq == 0 ? a0 : tq == 1 ? a1 : tq == 2 ? a2 : a3;
    int n = n0 + tq * 16 + nl;
    float ss = sc[n], bb = bi[n];
#pragma unroll
    for (int v = 0; v < 8; ++v) {
      float x = acc[v] * ss + bb;
      x = x > 0.f ? x : 0.f;
      int m = (mt << 4) + (h << 3) + v;
      if (store_mode == 0) {
        ((bf16*)outp)[(size_t)m * ldo + n] = (bf16)x;
      } else {
        int bb2 = m / rpb, r = m % rpb;
        ((float*)outp)[((size_t)bb2 * Nn + n) * (size_t)rpb + r] = x;
      }
    }
  }
}

// ---------------- SSG3 input builder: [l2f(256) | l2xyz(3) | pad] -> 288 ----
__global__ void build_ssg3_in(const bf16* __restrict__ l2f,
                              const float* __restrict__ l2xyz,
                              bf16* __restrict__ A, int rows) {
  int t = blockIdx.x * blockDim.x + threadIdx.x;
  if (t >= rows * 288) return;
  int m = t / 288, k = t % 288;
  bf16 v = (bf16)0.f;
  if (k < 256) v = l2f[(size_t)m * 256 + k];
  else if (k < 259) v = (bf16)l2xyz[(size_t)m * 3 + (k - 256)];
  A[t] = v;
}

// ---------------- 3-NN search (thread per unknown point) --------------------
// Known points staged in LDS; host picks blockDim so each block maps to a
// single batch (blockDim divides nu).
__global__ void three_nn(const float* __restrict__ unk, int ustride,
                         const float* __restrict__ kn, int kstride, int nu,
                         int nk, int* __restrict__ idx3,
                         float* __restrict__ w3, int total) {
  __shared__ float skx[512], sky[512], skz[512];
  int t = blockIdx.x * blockDim.x + threadIdx.x;
  int b = (blockIdx.x * blockDim.x) / nu;  // uniform per block
  const float* kp = kn + (size_t)b * nk * kstride;
  for (int i = threadIdx.x; i < nk; i += blockDim.x) {
    skx[i] = kp[(size_t)i * kstride + 0];
    sky[i] = kp[(size_t)i * kstride + 1];
    skz[i] = kp[(size_t)i * kstride + 2];
  }
  __syncthreads();
  if (t >= total) return;
  const float* up = unk + (size_t)t * ustride;
  float ux = up[0], uy = up[1], uz = up[2];
  float d0 = 1e30f, d1 = 1e30f, d2 = 1e30f;
  int i0 = 0, i1 = 0, i2 = 0;
  for (int j = 0; j < nk; ++j) {
    float dx = skx[j] - ux, dy = sky[j] - uy, dz = skz[j] - uz;
    float dd = dx * dx + dy * dy + dz * dz;
    if (dd < d0) { d2 = d1; i2 = i1; d1 = d0; i1 = i0; d0 = dd; i0 = j; }
    else if (dd < d1) { d2 = d1; i2 = i1; d1 = dd; i1 = j; }
    else if (dd < d2) { d2 = dd; i2 = j; }
  }
  float r0 = 1.f / (d0 + 1e-8f), r1 = 1.f / (d1 + 1e-8f), r2 = 1.f / (d2 + 1e-8f);
  float s = r0 + r1 + r2;
  idx3[(size_t)t * 3 + 0] = i0; idx3[(size_t)t * 3 + 1] = i1;
  idx3[(size_t)t * 3 + 2] = i2;
  w3[(size_t)t * 3 + 0] = r0 / s; w3[(size_t)t * 3 + 1] = r1 / s;
  w3[(size_t)t * 3 + 2] = r2 / s;
}

// ---------------- interpolate + concat skip -> padded bf16 GEMM input -------
__global__ void interp_concat(const bf16* __restrict__ feats, int C, int nk,
                              const void* __restrict__ skip, int Cs,
                              int skip_ld, int skip_f32,
                              const int* __restrict__ idx3,
                              const float* __restrict__ w3,
                              bf16* __restrict__ A, int Kp, int nu,
                              long long total) {
  long long t = (long long)blockIdx.x * blockDim.x + threadIdx.x;
  if (t >= total) return;
  int k = (int)(t % Kp);
  long long m = t / Kp;
  int b = (int)(m / nu);
  bf16 v = (bf16)0.f;
  if (k < C) {
    const int* id = idx3 + m * 3;
    const float* w = w3 + m * 3;
    const bf16* fb = feats + (size_t)b * nk * C;
    float x = w[0] * (float)fb[(size_t)id[0] * C + k] +
              w[1] * (float)fb[(size_t)id[1] * C + k] +
              w[2] * (float)fb[(size_t)id[2] * C + k];
    v = (bf16)x;
  } else if (k < C + Cs) {
    int kc = k - C;
    if (skip_f32) v = (bf16)((const float*)skip)[m * skip_ld + kc];
    else          v = ((const bf16*)skip)[m * skip_ld + kc];
  }
  A[t] = v;
}

// ============================ host orchestration ============================
extern "C" void kernel_launch(void* const* d_in, const int* in_sizes, int n_in,
                              void* d_out, int out_size, void* d_ws,
                              size_t ws_size, hipStream_t stream) {
  (void)in_sizes; (void)n_in; (void)out_size; (void)ws_size;
  const int B = 16, N = 8192, N1 = 512, N2 = 128;

  const float* pc = (const float*)d_in[0];
  // weights: (W,s,b) triples starting at index 2
  const float* Ws[16]; const float* Ss[16]; const float* Bs[16];
  {
    int widx[16] = {2, 5, 8, 11, 14, 17, 20, 23, 26, 29, 32, 35, 38, 41, 44, 47};
    for (int i = 0; i < 16; ++i) {
      Ws[i] = (const float*)d_in[widx[i]];
      Ss[i] = (const float*)d_in[widx[i] + 1];
      Bs[i] = (const float*)d_in[widx[i] + 2];
    }
  }
  // weight dims: cout, cin, kpad
  const int WD[16][3] = {
      {64, 6, 32},    {64, 64, 64},   {128, 64, 64},                 // ssg1
      {128, 131, 160},{128, 128, 128},{256, 128, 128},               // ssg2
      {256, 259, 288},{512, 256, 256},{1024, 512, 512},              // ssg3
      {256, 1280, 1280},{256, 256, 256},                             // fp1
      {256, 384, 384},{128, 256, 256},                               // fp2
      {128, 134, 160},{128, 128, 128},{128, 128, 128}};              // fp3

  size_t cur = 0;
  auto alloc = [&](size_t bytes) -> void* {
    void* p = (char*)d_ws + cur;
    cur += (bytes + 255) & ~(size_t)255;
    return p;
  };

  bf16* Wp[16];
  for (int i = 0; i < 16; ++i)
    Wp[i] = (bf16*)alloc((size_t)WD[i][0] * WD[i][2] * sizeof(bf16));

  int*   fidx1 = (int*)  alloc((size_t)B * N1 * sizeof(int));
  float* l1xyz = (float*)alloc((size_t)B * N1 * 3 * sizeof(float));
  int*   gidx1 = (int*)  alloc((size_t)B * N1 * 64 * sizeof(int));
  bf16*  l1f   = (bf16*) alloc((size_t)B * N1 * 128 * sizeof(bf16));
  int*   fidx2 = (int*)  alloc((size_t)B * N2 * sizeof(int));
  float* l2xyz = (float*)alloc((size_t)B * N2 * 3 * sizeof(float));
  int*   gidx2 = (int*)  alloc((size_t)B * N2 * 64 * sizeof(int));
  bf16*  l2f   = (bf16*) alloc((size_t)B * N2 * 256 * sizeof(bf16));
  bf16*  A3    = (bf16*) alloc((size_t)B * N2 * 288 * sizeof(bf16));
  bf16*  h3a   = (bf16*) alloc((size_t)B * N2 * 256 * sizeof(bf16));
  bf16*  h3b   = (bf16*) alloc((size_t)B * N2 * 512 * sizeof(bf16));
  bf16*  l3f   = (bf16*) alloc((size_t)B * N2 * 1024 * sizeof(bf16));
  int*   id1   = (int*)  alloc((size_t)B * N2 * 3 * sizeof(int));
  float* wt1   = (float*)alloc((size_t)B * N2 * 3 * sizeof(float));
  bf16*  Af1   = (bf16*) alloc((size_t)B * N2 * 1280 * sizeof(bf16));
  bf16*  hf1   = (bf16*) alloc((size_t)B * N2 * 256 * sizeof(bf16));
  bf16*  u1    = (bf16*) alloc((size_t)B * N2 * 256 * sizeof(bf16));
  int*   id2   = (int*)  alloc((size_t)B * N1 * 3 * sizeof(int));
  float* wt2   = (float*)alloc((size_t)B * N1 * 3 * sizeof(float));
  bf16*  Af2   = (bf16*) alloc((size_t)B * N1 * 384 * sizeof(bf16));
  bf16*  hf2   = (bf16*) alloc((size_t)B * N1 * 256 * sizeof(bf16));
  bf16*  u2    = (bf16*) alloc((size_t)B * N1 * 128 * sizeof(bf16));
  int*   id3   = (int*)  alloc((size_t)B * N * 3 * sizeof(int));
  float* wt3   = (float*)alloc((size_t)B * N * 3 * sizeof(float));
  bf16*  Af3   = (bf16*) alloc((size_t)B * N * 160 * sizeof(bf16));
  bf16*  hf3a  = (bf16*) alloc((size_t)B * N * 128 * sizeof(bf16));
  bf16*  hf3b  = (bf16*) alloc((size_t)B * N * 128 * sizeof(bf16));

  // --- pack all weights to padded bf16 ---
  for (int i = 0; i < 16; ++i) {
    int tot = WD[i][0] * WD[i][2];
    pack_weight<<<(tot + 255) / 256, 256, 0, stream>>>(Ws[i], Wp[i], WD[i][0],
                                                       WD[i][1], WD[i][2]);
  }

  auto gemm = [&](const bf16* A, int lda, int wi, void* out, int ldo, int M,
                  int Nn, int K, int mode, int rpb) {
    dim3 g((M / 16 + 7) / 8, Nn / 64);
    gemm_bf16_wmma<<<g, 256, 0, stream>>>(A, lda, Wp[wi], WD[wi][2], Ss[wi],
                                          Bs[wi], out, ldo, M, Nn, K, mode, rpb);
  };

  // ---- SA1 ----
  fps_kernel<<<B, 1024, (size_t)N * 3 * sizeof(float), stream>>>(pc, 6, N, N1,
                                                                 fidx1, l1xyz);
  ball_query<<<(B * N1 + 7) / 8, 256, (size_t)N * 3 * sizeof(float), stream>>>(
      pc, 6, l1xyz, B, N, N1, 0.2f * 0.2f, gidx1);
  sa1_fused<<<dim3(N1, B), 256, 0, stream>>>(pc, gidx1, l1xyz, Wp[0], Ss[0],
                                             Bs[0], Wp[1], Ss[1], Bs[1], Wp[2],
                                             Ss[2], Bs[2], l1f);
  // ---- SA2 ----
  fps_kernel<<<B, 1024, (size_t)N1 * 3 * sizeof(float), stream>>>(
      l1xyz, 3, N1, N2, fidx2, l2xyz);
  ball_query<<<(B * N2 + 7) / 8, 256, (size_t)N1 * 3 * sizeof(float), stream>>>(
      l1xyz, 3, l2xyz, B, N1, N2, 0.4f * 0.4f, gidx2);
  sa2_fused<<<dim3(N2, B), 256, 0, stream>>>(l1xyz, l1f, gidx2, l2xyz, Wp[3],
                                             Ss[3], Bs[3], Wp[4], Ss[4], Bs[4],
                                             Wp[5], Ss[5], Bs[5], l2f);
  // ---- SSG3 bottleneck MLP ----
  int M3 = B * N2;
  build_ssg3_in<<<(M3 * 288 + 255) / 256, 256, 0, stream>>>(l2f, l2xyz, A3, M3);
  gemm(A3, 288, 6, h3a, 256, M3, 256, 288, 0, 0);
  gemm(h3a, 256, 7, h3b, 512, M3, 512, 256, 0, 0);
  gemm(h3b, 512, 8, l3f, 1024, M3, 1024, 512, 0, 0);
  // ---- FP1 (l2 -> l2, self) ----
  three_nn<<<M3 / 128, 128, 0, stream>>>(l2xyz, 3, l2xyz, 3, N2, N2, id1, wt1,
                                         M3);
  {
    long long tot = (long long)M3 * 1280;
    interp_concat<<<(unsigned)((tot + 255) / 256), 256, 0, stream>>>(
        l3f, 1024, N2, l2f, 256, 256, 0, id1, wt1, Af1, 1280, N2, tot);
  }
  gemm(Af1, 1280, 9, hf1, 256, M3, 256, 1280, 0, 0);
  gemm(hf1, 256, 10, u1, 256, M3, 256, 256, 0, 0);
  // ---- FP2 (l2 -> l1) ----
  int M2 = B * N1;
  three_nn<<<M2 / 256, 256, 0, stream>>>(l1xyz, 3, l2xyz, 3, N1, N2, id2, wt2,
                                         M2);
  {
    long long tot = (long long)M2 * 384;
    interp_concat<<<(unsigned)((tot + 255) / 256), 256, 0, stream>>>(
        u1, 256, N2, l1f, 128, 128, 0, id2, wt2, Af2, 384, N1, tot);
  }
  gemm(Af2, 384, 11, hf2, 256, M2, 256, 384, 0, 0);
  gemm(hf2, 256, 12, u2, 128, M2, 128, 256, 0, 0);
  // ---- FP3 (l1 -> full cloud) ----
  int M1 = B * N;
  three_nn<<<M1 / 256, 256, 0, stream>>>(pc, 6, l1xyz, 3, N, N1, id3, wt3, M1);
  {
    long long tot = (long long)M1 * 160;
    interp_concat<<<(unsigned)((tot + 255) / 256), 256, 0, stream>>>(
        u2, 128, N1, pc, 6, 6, 1, id3, wt3, Af3, 160, N, tot);
  }
  gemm(Af3, 160, 13, hf3a, 128, M1, 128, 160, 0, 0);
  gemm(hf3a, 128, 14, hf3b, 128, M1, 128, 128, 0, 0);
  // final layer stores f32 transposed (B,128,8192) directly into d_out
  gemm(hf3b, 128, 15, d_out, 128, M1, 128, 128, 1, N);
}
